// GCMv2_59622736003126
// MI455X (gfx1250) — matrix operand order repaired and checked by reading, not compiled, not run on hardware
//
#include <hip/hip_runtime.h>
#include <stdint.h>

#define NN   3072
#define EE   49152
#define DD   128
#define GG   64
#define LL   4
#define HH   4
#define HDD  32
#define EXPD 256
#define NG   48                 // nodes per graph (contiguous, sorted batch)
#define ND   (NN * DD)

typedef __attribute__((ext_vector_type(16))) _Float16 v16h;
typedef __attribute__((ext_vector_type(8)))  float    v8f;
typedef int v4i32 __attribute__((vector_size(16)));

// 16-bit A-matrix fragment K index for element h of this lane (contiguous: k = 16*(lane/16) + h)
__device__ __forceinline__ int a_kidx(int h, int lane) {
  return ((lane >> 4) << 4) + h;
}

// ---- gfx1250 async global->LDS (ASYNCcnt protocol), with safe fallback ----
#if defined(__has_builtin)
#if __has_builtin(__builtin_amdgcn_global_load_async_to_lds_b128)
#define ASYNC_LDS 1
#endif
#endif
#ifndef ASYNC_LDS
#define ASYNC_LDS 0
#endif

__device__ __forceinline__ void g2l_b128(const float* g, float* l) {
#if ASYNC_LDS
  __builtin_amdgcn_global_load_async_to_lds_b128(
      (__attribute__((address_space(1))) v4i32*)(uintptr_t)g,
      (__attribute__((address_space(3))) v4i32*)(uintptr_t)l, 0, 0);
#else
  *(float4*)l = *(const float4*)g;
#endif
}
__device__ __forceinline__ void async_wait_all() {
#if ASYNC_LDS
#if __has_builtin(__builtin_amdgcn_s_wait_asynccnt)
  __builtin_amdgcn_s_wait_asynccnt(0);
#else
  asm volatile("s_wait_asynccnt 0x0" ::: "memory");
#endif
#endif
}

// ------------------------- elementwise helpers -------------------------
__global__ void k_fill0(float* __restrict__ p, int n) {
  int i = blockIdx.x * blockDim.x + threadIdx.x;
  if (i < n) p[i] = 0.0f;
}
__global__ void k_copy(float* __restrict__ d, const float* __restrict__ s, int n) {
  int i = blockIdx.x * blockDim.x + threadIdx.x;
  if (i < n) d[i] = s[i];
}
__global__ void k_add(float* __restrict__ h, const float* __restrict__ x,
                      const float* __restrict__ a, int n) {
  int i = blockIdx.x * blockDim.x + threadIdx.x;
  if (i < n) h[i] = x[i] + a[i];
}

// ------------------------- GENConv edge passes -------------------------
// msg = relu(x[src]+ea)+eps > 0  => fp32 segment-max via uint atomicMax
__global__ void k_edge_msgmax(const float* __restrict__ x, const float* __restrict__ ea,
                              const int* __restrict__ ei, float* __restrict__ mx) {
  int idx = blockIdx.x * blockDim.x + threadIdx.x;
  if (idx >= EE * DD) return;
  int e = idx >> 7, c = idx & (DD - 1);
  int s = ei[e], d = ei[EE + e];
  float m = fmaxf(x[(size_t)s * DD + c] + ea[idx], 0.0f) + 1e-7f;
  atomicMax((unsigned int*)&mx[(size_t)d * DD + c], __float_as_uint(m));
}
__global__ void k_edge_den(const float* __restrict__ x, const float* __restrict__ ea,
                           const int* __restrict__ ei, const float* __restrict__ mx,
                           float* __restrict__ den) {
  int idx = blockIdx.x * blockDim.x + threadIdx.x;
  if (idx >= EE * DD) return;
  int e = idx >> 7, c = idx & (DD - 1);
  int s = ei[e], d = ei[EE + e];
  float m = fmaxf(x[(size_t)s * DD + c] + ea[idx], 0.0f) + 1e-7f;
  size_t o = (size_t)d * DD + c;
  atomicAdd(&den[o], __expf(m - mx[o]));
}
__global__ void k_edge_aggr(const float* __restrict__ x, const float* __restrict__ ea,
                            const int* __restrict__ ei, const float* __restrict__ mx,
                            const float* __restrict__ den, float* __restrict__ aggr) {
  int idx = blockIdx.x * blockDim.x + threadIdx.x;
  if (idx >= EE * DD) return;
  int e = idx >> 7, c = idx & (DD - 1);
  int s = ei[e], d = ei[EE + e];
  float m = fmaxf(x[(size_t)s * DD + c] + ea[idx], 0.0f) + 1e-7f;
  size_t o = (size_t)d * DD + c;
  float ex = __expf(m - mx[o]);
  atomicAdd(&aggr[o], (ex / den[o]) * m);
}

// ------------------------- generic WMMA GEMM -------------------------
// C[M,N] = A[M,K] @ B[K,N] (+bias). 256 threads, 64x64 C-tile, each wave
// computes a 16x32 slab (A-frag reused over 2 WMMAs). K templated -> fully
// unrolled, LDS double-buffered: A staged raw f32 via async-to-LDS b128,
// B staged synchronously with f32->f16 convert + transpose.
template <int K>
__global__ __launch_bounds__(256) void k_gemm(const float* __restrict__ A,
    const float* __restrict__ B, const float* __restrict__ bias,
    float* __restrict__ C, int M, int N) {
  constexpr int STEPS = K / 32;
  __shared__ float    As[2][64][36];   // raw f32 [m][k], row stride 144B (16B aligned)
  __shared__ _Float16 Bt[2][64][40];   // f16 transposed [n][k]
  int tid = threadIdx.x, lane = tid & 31, wave = tid >> 5;
  int wm = wave & 3, wn = wave >> 2;
  int m0 = blockIdx.y * 64, n0 = blockIdx.x * 64;

  auto stageA = [&](int buf, int k0) {
#pragma unroll
    for (int i = 0; i < 2; ++i) {                // 64x32 f32 = 512 x b128 chunks
      int idx = i * 256 + tid;
      int m = idx >> 3, c4 = (idx & 7) << 2;
      g2l_b128(&A[(size_t)(m0 + m) * K + k0 + c4], &As[buf][m][c4]);
    }
  };
  auto stageB = [&](int buf, int k0) {
#pragma unroll
    for (int i = 0; i < 8; ++i) {                // 32x64, coalesced over n
      int idx = i * 256 + tid;
      int kk = idx >> 6, n = idx & 63;
      Bt[buf][n][kk] = (_Float16)B[(size_t)(k0 + kk) * N + n0 + n];
    }
  };

  stageA(0, 0);
  stageB(0, 0);
  v8f acc0 = {}, acc1 = {};
  int mrow = (wm << 4) + (lane & 15);
  int nc0 = (wn << 5) + (lane & 15);
  int kb = (lane >> 4) << 4;
#pragma unroll
  for (int s = 0; s < STEPS; ++s) {
    int buf = s & 1;
    async_wait_all();
    __syncthreads();                 // staged tile visible; prev buffer free
    if (s + 1 < STEPS) { stageA(buf ^ 1, (s + 1) * 32); stageB(buf ^ 1, (s + 1) * 32); }
    v16h af, b0, b1;
#pragma unroll
    for (int h = 0; h < 16; ++h) af[h] = (_Float16)As[buf][mrow][a_kidx(h, lane)];
#pragma unroll
    for (int h = 0; h < 16; ++h) b0[h] = Bt[buf][nc0][kb + h];
#pragma unroll
    for (int h = 0; h < 16; ++h) b1[h] = Bt[buf][nc0 + 16][kb + h];
    acc0 = __builtin_amdgcn_wmma_f32_16x16x32_f16(false, af, false, b0,
                                                  (short)0, acc0, false, false);
    acc1 = __builtin_amdgcn_wmma_f32_16x16x32_f16(false, af, false, b1,
                                                  (short)0, acc1, false, false);
  }
  int r0 = m0 + (wm << 4) + ((lane >> 4) << 3);
  int c0 = n0 + (wn << 5) + (lane & 15);
  float bv0 = bias ? bias[c0] : 0.f;
  float bv1 = bias ? bias[c0 + 16] : 0.f;
#pragma unroll
  for (int r = 0; r < 8; ++r) {
    C[(size_t)(r0 + r) * N + c0]      = acc0[r] + bv0;
    C[(size_t)(r0 + r) * N + c0 + 16] = acc1[r] + bv1;
  }
}

// ------------------------- BatchNorm (train mode, over N rows) -------------------------
__global__ __launch_bounds__(256) void k_bn_stats(const float* __restrict__ h1,
    float* __restrict__ mean, float* __restrict__ rstd) {
  int c = blockIdx.x * 32 + (threadIdx.x & 31);
  int rg = threadIdx.x >> 5;
  float s = 0.f, s2 = 0.f;
  for (int r = rg; r < NN; r += 8) {
    float v = h1[(size_t)r * EXPD + c];
    s += v; s2 += v * v;
  }
  __shared__ float ss[8][32], qq[8][32];
  ss[rg][threadIdx.x & 31] = s; qq[rg][threadIdx.x & 31] = s2;
  __syncthreads();
  if (rg == 0) {
    int ln = threadIdx.x & 31;
    float t = 0.f, t2 = 0.f;
    for (int i = 0; i < 8; ++i) { t += ss[i][ln]; t2 += qq[i][ln]; }
    float mu = t / (float)NN;
    mean[c] = mu;
    rstd[c] = rsqrtf(t2 / (float)NN - mu * mu + 1e-5f);
  }
}
__global__ void k_bn_relu(float* __restrict__ h1, const float* __restrict__ mean,
    const float* __restrict__ rstd, const float* __restrict__ g,
    const float* __restrict__ b) {
  int i = blockIdx.x * blockDim.x + threadIdx.x;
  if (i >= NN * EXPD) return;
  int c = i & (EXPD - 1);
  float v = (h1[i] - mean[c]) * rstd[c] * g[c] + b[c];
  h1[i] = fmaxf(v, 0.f);
}

// ------------------------- block-diagonal masked MHA -------------------------
// one block per (graph, head): 48x48 scores + softmax + P@V, all WMMA in LDS.
// masked inter-graph entries contribute exp(-1e9 - m) == 0 exactly, so this
// block-diagonal computation equals the reference's dense masked softmax.
__global__ __launch_bounds__(128) void k_attn(const float* __restrict__ Q,
    const float* __restrict__ Kb, const float* __restrict__ Vb,
    float* __restrict__ O) {
  __shared__ _Float16 Qs[NG][40];   // [m][k]
  __shared__ _Float16 Ks[NG][40];   // [j][d] -> B[d][j] frag reads contiguous
  __shared__ _Float16 Vt[HDD][72];  // transposed V: [n][k], k padded 48->64 (zeros)
  __shared__ float    Sf[NG][52];
  __shared__ _Float16 Pm[NG][72];   // P as A-matrix, cols 48..63 zero
  int g  = blockIdx.x >> 2;
  int hd = blockIdx.x & 3;
  int tid = threadIdx.x, lane = tid & 31, wave = tid >> 5;
  int base = g * NG;
  int co = hd * HDD;
#pragma unroll
  for (int i = 0; i < 12; ++i) {    // 48x32 Q and K tiles
    int idx = i * 128 + tid;
    int m = idx >> 5, kk = idx & 31;
    Qs[m][kk] = (_Float16)Q[(size_t)(base + m) * DD + co + kk];
    Ks[m][kk] = (_Float16)Kb[(size_t)(base + m) * DD + co + kk];
  }
#pragma unroll
  for (int i = 0; i < 16; ++i) {    // V transposed + zero pad K dim to 64
    int idx = i * 128 + tid;
    int n = idx >> 6, kk = idx & 63;
    Vt[n][kk] = (kk < NG) ? (_Float16)Vb[(size_t)(base + kk) * DD + co + n]
                          : (_Float16)0.f;
  }
  __syncthreads();
  const float scale = 0.17677669529663687f;   // 1/sqrt(32)
  for (int t = wave; t < 9; t += 4) {         // 3x3 score tiles
    int tm = t / 3, tn = t % 3;
    v16h af, bfr;
#pragma unroll
    for (int h = 0; h < 16; ++h) af[h] = Qs[tm * 16 + (lane & 15)][a_kidx(h, lane)];
    int kb = (lane >> 4) << 4;
#pragma unroll
    for (int h = 0; h < 16; ++h) bfr[h] = Ks[tn * 16 + (lane & 15)][kb + h];
    v8f acc = {};
    acc = __builtin_amdgcn_wmma_f32_16x16x32_f16(false, af, false, bfr,
                                                 (short)0, acc, false, false);
    int r0 = tm * 16 + ((lane >> 4) << 3);
#pragma unroll
    for (int r = 0; r < 8; ++r)
      Sf[r0 + r][tn * 16 + (lane & 15)] = acc[r] * scale;
  }
  __syncthreads();
  if (tid < NG) {                             // full-row softmax (row fits in block)
    float m = -1e30f;
    for (int j = 0; j < NG; ++j) m = fmaxf(m, Sf[tid][j]);
    float s = 0.f;
    for (int j = 0; j < NG; ++j) s += __expf(Sf[tid][j] - m);
    float inv = 1.f / s;
    for (int j = 0; j < 64; ++j)
      Pm[tid][j] = (j < NG) ? (_Float16)(__expf(Sf[tid][j] - m) * inv)
                            : (_Float16)0.f;
  }
  __syncthreads();
  for (int t = wave; t < 6; t += 4) {         // 3x2 output tiles, K = 64 (2 steps)
    int tm = t >> 1, tn = t & 1;
    v8f acc = {};
#pragma unroll
    for (int ks = 0; ks < 2; ++ks) {
      v16h af, bfr;
      int kk2 = ks * 32;
#pragma unroll
      for (int h = 0; h < 16; ++h)
        af[h] = Pm[tm * 16 + (lane & 15)][kk2 + a_kidx(h, lane)];
      int kb = (lane >> 4) << 4;
#pragma unroll
      for (int h = 0; h < 16; ++h)
        bfr[h] = Vt[tn * 16 + (lane & 15)][kk2 + kb + h];
      acc = __builtin_amdgcn_wmma_f32_16x16x32_f16(false, af, false, bfr,
                                                   (short)0, acc, false, false);
    }
    int r0 = tm * 16 + ((lane >> 4) << 3);
    int col = co + tn * 16 + (lane & 15);
#pragma unroll
    for (int r = 0; r < 8; ++r)
      O[(size_t)(base + r0 + r) * DD + col] = acc[r];
  }
}

// ------------------------- LayerNorm + private residual -------------------------
__global__ __launch_bounds__(128) void k_ln_pri(const float* __restrict__ tmp,
    const float* __restrict__ x, const float* __restrict__ g,
    const float* __restrict__ b, float* __restrict__ com, float* __restrict__ pri) {
  int row = blockIdx.x, t = threadIdx.x;
  __shared__ float red[128];
  __shared__ float mu_s, var_s;
  float v = tmp[(size_t)row * DD + t];
  red[t] = v; __syncthreads();
  for (int s = 64; s > 0; s >>= 1) { if (t < s) red[t] += red[t + s]; __syncthreads(); }
  if (t == 0) mu_s = red[0] * (1.f / DD);
  __syncthreads();
  float d = v - mu_s;
  red[t] = d * d; __syncthreads();
  for (int s = 64; s > 0; s >>= 1) { if (t < s) red[t] += red[t + s]; __syncthreads(); }
  if (t == 0) var_s = red[0] * (1.f / DD);
  __syncthreads();
  float c = d * rsqrtf(var_s + 1e-5f) * g[t] + b[t];
  com[(size_t)row * DD + t] = c;
  pri[(size_t)row * DD + t] = x[(size_t)row * DD + t] - c;
}

// ------------------------- attention fusion -------------------------
__global__ __launch_bounds__(128) void k_fuse_logits(const float* __restrict__ base,
    const float* __restrict__ Wf, const float* __restrict__ bfp, float* __restrict__ z) {
  int n = blockIdx.x, l = blockIdx.y, t = threadIdx.x;
  __shared__ float red[128];
  red[t] = base[(size_t)l * ND + (size_t)n * DD + t] * Wf[t];
  __syncthreads();
  for (int s = 64; s > 0; s >>= 1) { if (t < s) red[t] += red[t + s]; __syncthreads(); }
  if (t == 0) z[l * NN + n] = red[0] + bfp[0];
}
__global__ __launch_bounds__(256) void k_fuse_softmax(const float* __restrict__ z,
    float* __restrict__ smax, float* __restrict__ ssum) {
  int l = blockIdx.x, t = threadIdx.x;
  __shared__ float red[256];
  __shared__ float mm;
  float m = -1e30f;
  for (int n = t; n < NN; n += 256) m = fmaxf(m, z[l * NN + n]);
  red[t] = m; __syncthreads();
  for (int s = 128; s > 0; s >>= 1) { if (t < s) red[t] = fmaxf(red[t], red[t + s]); __syncthreads(); }
  if (t == 0) mm = red[0];
  __syncthreads();
  float s = 0.f;
  for (int n = t; n < NN; n += 256) s += __expf(z[l * NN + n] - mm);
  red[t] = s; __syncthreads();
  for (int q = 128; q > 0; q >>= 1) { if (t < q) red[t] += red[t + q]; __syncthreads(); }
  if (t == 0) { smax[l] = mm; ssum[l] = red[0]; }
}
__global__ void k_fuse_combine(const float* __restrict__ base, const float* __restrict__ z,
    const float* __restrict__ smax, const float* __restrict__ ssum,
    float* __restrict__ out) {
  int i = blockIdx.x * blockDim.x + threadIdx.x;
  if (i >= ND) return;
  int n = i >> 7;
  float w[LL], tw = 0.f;
#pragma unroll
  for (int l = 0; l < LL; ++l) { w[l] = __expf(z[l * NN + n] - smax[l]) / ssum[l]; tw += w[l]; }
  float acc = 0.f;
#pragma unroll
  for (int l = 0; l < LL; ++l) acc += w[l] * base[(size_t)l * ND + i];
  out[i] = acc / tw;
}

// ------------------------- gated global attention pool -------------------------
__global__ __launch_bounds__(128) void k_pool(const float* __restrict__ x,
    const float* __restrict__ Wg, const float* __restrict__ bgp,
    float* __restrict__ combined, int off) {
  int g = blockIdx.x, t = threadIdx.x;
  __shared__ float an[NG];
  if (t < NG) {
    const float* xp = x + (size_t)(g * NG + t) * DD;
    float s = 0.f;
    for (int c = 0; c < DD; ++c) s += xp[c] * Wg[c];
    an[t] = 1.f / (1.f + __expf(-(s + bgp[0])));
  }
  __syncthreads();
  if (t == 0) {
    float m = -1e30f;
    for (int n = 0; n < NG; ++n) m = fmaxf(m, an[n]);
    float s = 0.f;
    for (int n = 0; n < NG; ++n) { an[n] = __expf(an[n] - m); s += an[n]; }
    float inv = 1.f / s;
    for (int n = 0; n < NG; ++n) an[n] *= inv;
  }
  __syncthreads();
  float acc = 0.f;
  for (int n = 0; n < NG; ++n) acc += an[n] * x[(size_t)(g * NG + n) * DD + t];
  combined[(size_t)g * 4 * DD + off + t] = acc;
}

// ------------------------- classifier head -------------------------
__global__ void k_head(const float* __restrict__ comb, const float* __restrict__ Wc,
                       const float* __restrict__ bcp, float* __restrict__ out) {
  int g = threadIdx.x;
  if (g >= GG) return;
  const float* cp = comb + (size_t)g * (4 * DD);
  float z0 = bcp[0], z1 = bcp[1];
  for (int j = 0; j < 4 * DD; ++j) { z0 += cp[j] * Wc[j * 2]; z1 += cp[j] * Wc[j * 2 + 1]; }
  float m = fmaxf(z0, z1);
  float e0 = __expf(z0 - m), e1 = __expf(z1 - m), s = e0 + e1;
  out[g * 2] = e0 / s; out[g * 2 + 1] = e1 / s;
}

// ------------------------- host orchestration -------------------------
extern "C" void kernel_launch(void* const* d_in, const int* in_sizes, int n_in,
                              void* d_out, int out_size, void* d_ws, size_t ws_size,
                              hipStream_t stream) {
  (void)in_sizes; (void)n_in; (void)out_size; (void)ws_size;
  const float* x_source = (const float*)d_in[0];
  const int*   ei_s     = (const int*)  d_in[1];
  const float* ea_s     = (const float*)d_in[2];
  const float* x_target = (const float*)d_in[4];
  const int*   ei_t     = (const int*)  d_in[5];
  const float* ea_t     = (const float*)d_in[6];
  const float* W1 = (const float*)d_in[8];
  const float* g1 = (const float*)d_in[9];
  const float* b1 = (const float*)d_in[10];
  const float* W2 = (const float*)d_in[11];
  const float* Wq = (const float*)d_in[12]; const float* bq = (const float*)d_in[13];
  const float* Wk = (const float*)d_in[14]; const float* bk = (const float*)d_in[15];
  const float* Wv = (const float*)d_in[16]; const float* bv = (const float*)d_in[17];
  const float* Wo = (const float*)d_in[18]; const float* bo = (const float*)d_in[19];
  const float* lng = (const float*)d_in[20]; const float* lnb = (const float*)d_in[21];
  const float* Wf = (const float*)d_in[22]; const float* bfp = (const float*)d_in[23];
  const float* Wg = (const float*)d_in[24]; const float* bgp = (const float*)d_in[25];
  const float* Wc = (const float*)d_in[26]; const float* bcp = (const float*)d_in[27];

  float* ws = (float*)d_ws;
  size_t off = 0;
  auto WS = [&](size_t n) { float* p = ws + off; off += n; return p; };
  float* mx   = WS(ND);
  float* den  = WS(ND);
  float* aggr = WS(ND);
  float* hb   = WS(ND);
  float* h1   = WS((size_t)NN * EXPD);
  float* bnm  = WS(512);
  float* bnr  = WS(512);
  float* xs   = WS(ND);
  float* xt   = WS(ND);
  float* Qs_  = WS(ND); float* Ks_ = WS(ND); float* Vs_ = WS(ND);
  float* Qt_  = WS(ND); float* Kt_ = WS(ND); float* Vt_ = WS(ND);
  float* as_  = WS(ND); float* at_ = WS(ND);
  float* ts_  = WS(ND); float* tt_ = WS(ND);
  float* com_s = WS((size_t)LL * ND); float* com_t = WS((size_t)LL * ND);
  float* pri_s = WS((size_t)LL * ND); float* pri_t = WS((size_t)LL * ND);
  float* fusebuf = WS(ND);
  float* z    = WS(LL * NN);
  float* smax = WS(16);
  float* ssum = WS(16);
  float* comb = WS((size_t)GG * 4 * DD);

  const int gED = (EE * DD + 255) / 256;
  const int gND = (ND + 255) / 256;
  auto gemm128 = [&](const float* Aa, const float* Bb, const float* bias, float* Cc,
                     int M, int N2) {
    k_gemm<128><<<dim3(N2 / 64, M / 64), 256, 0, stream>>>(Aa, Bb, bias, Cc, M, N2);
  };
  auto gemm256 = [&](const float* Aa, const float* Bb, const float* bias, float* Cc,
                     int M, int N2) {
    k_gemm<256><<<dim3(N2 / 64, M / 64), 256, 0, stream>>>(Aa, Bb, bias, Cc, M, N2);
  };

  k_copy<<<gND, 256, 0, stream>>>(xs, x_source, ND);
  k_copy<<<gND, 256, 0, stream>>>(xt, x_target, ND);

  for (int i = 0; i < LL; ++i) {
    const float* W1i = W1 + (size_t)i * DD * EXPD;
    const float* g1i = g1 + (size_t)i * EXPD;
    const float* b1i = b1 + (size_t)i * EXPD;
    const float* W2i = W2 + (size_t)i * EXPD * DD;
    for (int side = 0; side < 2; ++side) {
      float* x = side ? xt : xs;
      const int* ei = side ? ei_t : ei_s;
      const float* ea = side ? ea_t : ea_s;
      k_fill0<<<gND, 256, 0, stream>>>(mx, ND);
      k_fill0<<<gND, 256, 0, stream>>>(den, ND);
      k_fill0<<<gND, 256, 0, stream>>>(aggr, ND);
      k_edge_msgmax<<<gED, 256, 0, stream>>>(x, ea, ei, mx);
      k_edge_den<<<gED, 256, 0, stream>>>(x, ea, ei, mx, den);
      k_edge_aggr<<<gED, 256, 0, stream>>>(x, ea, ei, mx, den, aggr);
      k_add<<<gND, 256, 0, stream>>>(hb, x, aggr, ND);
      gemm128(hb, W1i, nullptr, h1, NN, EXPD);
      k_bn_stats<<<EXPD / 32, 256, 0, stream>>>(h1, bnm, bnr);
      k_bn_relu<<<(NN * EXPD + 255) / 256, 256, 0, stream>>>(h1, bnm, bnr, g1i, b1i);
      gemm256(h1, W2i, nullptr, x, NN, DD);
    }
    const float* Wqi = Wq + (size_t)i * DD * DD; const float* bqi = bq + (size_t)i * DD;
    const float* Wki = Wk + (size_t)i * DD * DD; const float* bki = bk + (size_t)i * DD;
    const float* Wvi = Wv + (size_t)i * DD * DD; const float* bvi = bv + (size_t)i * DD;
    const float* Woi = Wo + (size_t)i * DD * DD; const float* boi = bo + (size_t)i * DD;
    gemm128(xs, Wqi, bqi, Qs_, NN, DD);
    gemm128(xs, Wki, bki, Ks_, NN, DD);
    gemm128(xs, Wvi, bvi, Vs_, NN, DD);
    gemm128(xt, Wqi, bqi, Qt_, NN, DD);
    gemm128(xt, Wki, bki, Kt_, NN, DD);
    gemm128(xt, Wvi, bvi, Vt_, NN, DD);
    k_attn<<<GG * HH, 128, 0, stream>>>(Qs_, Kt_, Vt_, as_);   // src queries, tgt keys/vals
    k_attn<<<GG * HH, 128, 0, stream>>>(Qt_, Ks_, Vs_, at_);   // tgt queries, src keys/vals
    gemm128(as_, Woi, boi, ts_, NN, DD);
    gemm128(at_, Woi, boi, tt_, NN, DD);
    k_ln_pri<<<NN, 128, 0, stream>>>(ts_, xs, lng + (size_t)i * DD, lnb + (size_t)i * DD,
                                     com_s + (size_t)i * ND, pri_s + (size_t)i * ND);
    k_ln_pri<<<NN, 128, 0, stream>>>(tt_, xt, lng + (size_t)i * DD, lnb + (size_t)i * DD,
                                     com_t + (size_t)i * ND, pri_t + (size_t)i * ND);
  }

  // combined = [p_cs | p_ps | p_ct | p_pt]
  const float* stacks[4] = {com_s, pri_s, com_t, pri_t};
  const int offs[4] = {0, DD, 2 * DD, 3 * DD};
  for (int f = 0; f < 4; ++f) {
    dim3 gl(NN, LL);
    k_fuse_logits<<<gl, 128, 0, stream>>>(stacks[f], Wf, bfp, z);
    k_fuse_softmax<<<LL, 256, 0, stream>>>(z, smax, ssum);
    k_fuse_combine<<<gND, 256, 0, stream>>>(stacks[f], z, smax, ssum, fusebuf);
    k_pool<<<GG, 128, 0, stream>>>(fusebuf, Wg, bgp, comb, offs[f]);
  }
  k_head<<<1, 64, 0, stream>>>(comb, Wc, bcp, (float*)d_out);
}
